// FEDForecaster_17841294148198
// MI455X (gfx1250) — matrix operand order repaired and checked by reading, not compiled, not run on hardware
//
#include <hip/hip_runtime.h>
#include <math.h>

typedef __attribute__((ext_vector_type(16))) _Float16 v16h;
typedef __attribute__((ext_vector_type(8)))  float    v8f;

constexpr int CB    = 16;    // batch
constexpr int CT    = 1024;  // input time
constexpr int CFIN  = 64;
constexpr int CD    = 512;
constexpr int CNH   = 8;
constexpr int CDH   = 64;
constexpr int CDFF  = 2048;
constexpr int CTP   = 1048;  // T + KSIZE - 1
constexpr int CFQ   = 525;   // TP/2 + 1
constexpr int CTOPK = 16;
constexpr int COUTN = 192;   // 24*8

enum { GF_BIAS = 1, GF_RELU = 2, GF_RESID = 4 };

// ---------------------------------------------------------------------------
// Generic WMMA GEMM: C[b] = A[b] * B[b] (+bias)(+resid)(ReLU), fp32 I/O,
// f16 MACs with fp32 accumulation via v_wmma_f32_16x16x32_f16.
// Block tile 128x64, BK=32, 256 threads = 8 waves, wave tile 32x32 (2x2 WMMA).
// ---------------------------------------------------------------------------
#define TBM 128
#define TBN 64
#define TBK 32
#define LSTR 40

template<int FLAGS>
__global__ __launch_bounds__(256)
void gemm_wmma_k(const float* __restrict__ A, const float* __restrict__ Bm,
                 float* __restrict__ C, const float* __restrict__ bias,
                 const float* __restrict__ resid,
                 int M, int N, int K, long sA, long sB, long sC, long sR)
{
  __shared__ _Float16 As[TBM * LSTR];
  __shared__ _Float16 Bs[TBN * LSTR];
  const int tid  = threadIdx.x;
  const int lane = tid & 31;
  const int wave = tid >> 5;
  const int wm   = wave & 3;   // 4 wave-rows of 32
  const int wn   = wave >> 2;  // 2 wave-cols of 32
  const int kg   = lane >> 4;
  const int ln16 = lane & 15;
  const int bz = blockIdx.z;
  A += (long)bz * sA; Bm += (long)bz * sB; C += (long)bz * sC;
  const float* rz = resid;
  if (FLAGS & GF_RESID) rz += (long)bz * sR;
  const int m0 = blockIdx.y * TBM;
  const int n0 = blockIdx.x * TBN;

  v8f acc[2][2] = {};

  for (int k0 = 0; k0 < K; k0 += TBK) {
    // stage A (128x32) -> LDS f16
#pragma unroll
    for (int i = 0; i < 16; ++i) {
      int idx = i * 256 + tid;
      int m = idx >> 5, k = idx & 31;
      int gm = m0 + m, gk = k0 + k;
      float v = (gm < M && gk < K) ? A[(long)gm * K + gk] : 0.f;
      As[m * LSTR + k] = (_Float16)v;
    }
    // stage B (32x64) transposed -> Bs[n][k]
#pragma unroll
    for (int i = 0; i < 8; ++i) {
      int idx = i * 256 + tid;
      int k = idx >> 6, n = idx & 63;
      int gk = k0 + k, gn = n0 + n;
      float v = (gk < K && gn < N) ? Bm[(long)gk * N + gn] : 0.f;
      Bs[n * LSTR + k] = (_Float16)v;
    }
    __syncthreads();

    v16h af[2], bf[2];
#pragma unroll
    for (int s = 0; s < 2; ++s) {
      const _Float16* ap = &As[(wm * 32 + s * 16 + ln16) * LSTR];
#pragma unroll
      for (int i = 0; i < 16; ++i) {          // A layout: k = 16*kg+i
        int k = 16 * kg + i;
        af[s][i] = ap[k];
      }
      const _Float16* bp = &Bs[(wn * 32 + s * 16 + ln16) * LSTR + 16 * kg];
#pragma unroll
      for (int i = 0; i < 16; ++i) bf[s][i] = bp[i];  // B layout: k = 16*kg+i
    }
#pragma unroll
    for (int sm = 0; sm < 2; ++sm)
#pragma unroll
      for (int sn = 0; sn < 2; ++sn)
        acc[sm][sn] = __builtin_amdgcn_wmma_f32_16x16x32_f16(
            false, af[sm], false, bf[sn], (short)0, acc[sm][sn], false, false);
    __syncthreads();
  }

#pragma unroll
  for (int sm = 0; sm < 2; ++sm)
#pragma unroll
    for (int sn = 0; sn < 2; ++sn) {
      int n = n0 + wn * 32 + sn * 16 + ln16;
#pragma unroll
      for (int r = 0; r < 8; ++r) {
        int m = m0 + wm * 32 + sm * 16 + r + 8 * kg;
        if (m < M && n < N) {
          float v = acc[sm][sn][r];
          if (FLAGS & GF_BIAS)  v += bias[n];
          if (FLAGS & GF_RESID) v += rz[(long)m * N + n];
          if (FLAGS & GF_RELU)  v = v > 0.f ? v : 0.f;
          C[(long)m * N + n] = v;
        }
      }
    }
}

// ---------------------------------------------------------------------------
// Flash attention over qkv buffer (NT x 1536): per wave one 16-row query tile,
// 32-key tiles, online softmax, all contractions on WMMA.
// ---------------------------------------------------------------------------
#define PPAD 36
__global__ __launch_bounds__(256)
void flash_attn_k(const float* __restrict__ qkv, float* __restrict__ out)
{
  __shared__ _Float16 Pl[8][16 * PPAD];
  const int tid = threadIdx.x, lane = tid & 31, wave = tid >> 5;
  const int kg = lane >> 4, ln16 = lane & 15;
  const int bh = blockIdx.x, b = bh >> 3, hh = bh & 7;
  const int qt = blockIdx.y * 8 + wave;
  const int q0 = qt * 16;
  const int mr = q0 + ln16;
  const bool qv = mr < CTP;
  const long baseQ = ((long)(b * CTP + (qv ? mr : 0))) * (3 * CD) + hh * CDH;

  v16h aq[2];
#pragma unroll
  for (int s = 0; s < 2; ++s)
#pragma unroll
    for (int i = 0; i < 16; ++i) {
      int k = 16 * kg + i + 32 * s;
      aq[s][i] = qv ? (_Float16)qkv[baseQ + k] : (_Float16)0.f;
    }

  float mrow[8], lrow[8];
#pragma unroll
  for (int r = 0; r < 8; ++r) { mrow[r] = -3.0e38f; lrow[r] = 0.f; }
  v8f o[4] = {};

  const int nkt = (CTP + 31) / 32;   // 33
  for (int jt = 0; jt < nkt; ++jt) {
    const int j0 = jt * 32;
    v8f sa[2];
#pragma unroll
    for (int c = 0; c < 2; ++c) {
      int key = j0 + c * 16 + ln16;
      bool kv = key < CTP;
      long baseK = ((long)(b * CTP + (kv ? key : 0))) * (3 * CD) + CD + hh * CDH;
      v16h kf0, kf1;
#pragma unroll
      for (int i = 0; i < 16; ++i) {
        int k = 16 * kg + i;
        kf0[i] = kv ? (_Float16)qkv[baseK + k]      : (_Float16)0.f;
        kf1[i] = kv ? (_Float16)qkv[baseK + 32 + k] : (_Float16)0.f;
      }
      v8f s_ = {};
      s_ = __builtin_amdgcn_wmma_f32_16x16x32_f16(false, aq[0], false, kf0, (short)0, s_, false, false);
      s_ = __builtin_amdgcn_wmma_f32_16x16x32_f16(false, aq[1], false, kf1, (short)0, s_, false, false);
      sa[c] = s_;
    }
    float p0[8], p1[8];
    const int key0 = j0 + ln16, key1 = j0 + 16 + ln16;
#pragma unroll
    for (int r = 0; r < 8; ++r) {
      float s0 = (key0 < CTP) ? sa[0][r] * 0.125f : -3.0e38f;
      float s1 = (key1 < CTP) ? sa[1][r] * 0.125f : -3.0e38f;
      float mx = fmaxf(s0, s1);
#pragma unroll
      for (int msk = 1; msk < 16; msk <<= 1) mx = fmaxf(mx, __shfl_xor(mx, msk, 32));
      float mnew = fmaxf(mrow[r], mx);
      float scl = __expf(mrow[r] - mnew);
      float e0 = __expf(s0 - mnew), e1 = __expf(s1 - mnew);
      float ps = e0 + e1;
#pragma unroll
      for (int msk = 1; msk < 16; msk <<= 1) ps += __shfl_xor(ps, msk, 32);
      lrow[r] = lrow[r] * scl + ps;
      mrow[r] = mnew;
      o[0][r] *= scl; o[1][r] *= scl; o[2][r] *= scl; o[3][r] *= scl;
      p0[r] = e0; p1[r] = e1;
    }
    // transpose P (C-layout -> A-layout) through this wave's LDS patch
#pragma unroll
    for (int r = 0; r < 8; ++r) {
      Pl[wave][(r + 8 * kg) * PPAD + ln16]      = (_Float16)p0[r];
      Pl[wave][(r + 8 * kg) * PPAD + 16 + ln16] = (_Float16)p1[r];
    }
    __syncthreads();
    v16h pa;
#pragma unroll
    for (int i = 0; i < 16; ++i) {
      int k = 16 * kg + i;
      pa[i] = Pl[wave][ln16 * PPAD + k];
    }
    __syncthreads();
#pragma unroll
    for (int nt = 0; nt < 4; ++nt) {
      v16h vf;
#pragma unroll
      for (int i = 0; i < 16; ++i) {
        int k = 16 * kg + i;
        int key = j0 + k;
        vf[i] = (key < CTP)
          ? (_Float16)qkv[((long)(b * CTP + key)) * (3 * CD) + 2 * CD + hh * CDH + nt * 16 + ln16]
          : (_Float16)0.f;
      }
      o[nt] = __builtin_amdgcn_wmma_f32_16x16x32_f16(false, pa, false, vf, (short)0, o[nt], false, false);
    }
  }
#pragma unroll
  for (int r = 0; r < 8; ++r) {
    int m = q0 + r + 8 * kg;
    if (m < CTP) {
      float inv = 1.f / lrow[r];
      long baseO = ((long)(b * CTP + m)) * CD + hh * CDH + ln16;
      out[baseO + 0]  = o[0][r] * inv;
      out[baseO + 16] = o[1][r] * inv;
      out[baseO + 32] = o[2][r] * inv;
      out[baseO + 48] = o[3][r] * inv;
    }
  }
}

// --------------------------- pointwise helpers -----------------------------
__global__ void decompose_k(const float* __restrict__ x, float* __restrict__ trend,
                            float* __restrict__ seas)
{
  long i = (long)blockIdx.x * 256 + threadIdx.x;
  const long total = (long)CB * CTP * CFIN;
  if (i >= total) return;
  int f = (int)(i % CFIN); long r = i / CFIN;
  int t = (int)(r % CTP);  int b = (int)(r / CTP);
  const float* xb = x + (long)b * CT * CFIN + f;
  float s = 0.f;
  for (int j = 0; j < 25; ++j) {
    int u = t + j - 24;
    if (u >= 0 && u < CT) s += xb[(long)u * CFIN];
  }
  float tr = s * (1.f / 25.f);
  int v = t - 12;
  float xs = (v >= 0 && v < CT) ? xb[(long)v * CFIN] : 0.f;
  trend[i] = tr; seas[i] = xs - tr;
}

__global__ void posenc_add_k(float* __restrict__ h)
{
  long i = (long)blockIdx.x * 256 + threadIdx.x;
  const long total = (long)CB * CTP * CD;
  if (i >= total) return;
  int d = (int)(i % CD); int t = (int)((i / CD) % CTP);
  int p = d >> 1;
  float div = __expf(-(float)(2 * p) * (9.210340371976184f / (float)CD));
  float ang = (float)t * div;
  h[i] += (d & 1) ? cosf(ang) : sinf(ang);
}

__global__ void twiddle_k(float* __restrict__ cw, float* __restrict__ sw)
{
  long i = (long)blockIdx.x * 256 + threadIdx.x;
  const long total = (long)CFQ * CTP;
  if (i >= total) return;
  int f = (int)(i / CTP), t = (int)(i % CTP);
  int ph = (int)(((long)f * t) % CTP);
  float ang = 6.283185307179586f * (float)ph / (float)CTP;
  cw[i] = cosf(ang);
  sw[i] = -sinf(ang);   // rfft: Im = -sum x sin
}

__global__ __launch_bounds__(256)
void ln_k(const float* __restrict__ x, const float* __restrict__ g,
          const float* __restrict__ be, float* __restrict__ y)
{
  long row = blockIdx.x;
  const float* xr = x + row * (long)CD;
  int tid = threadIdx.x;
  float v0 = xr[tid], v1 = xr[tid + 256];
  __shared__ float red[256];
  __shared__ float smu, sinv;
  red[tid] = v0 + v1; __syncthreads();
  for (int st = 128; st > 0; st >>= 1) { if (tid < st) red[tid] += red[tid + st]; __syncthreads(); }
  if (tid == 0) smu = red[0] / (float)CD;
  __syncthreads();
  float mu = smu;
  float a0 = v0 - mu, a1 = v1 - mu;
  red[tid] = a0 * a0 + a1 * a1; __syncthreads();
  for (int st = 128; st > 0; st >>= 1) { if (tid < st) red[tid] += red[tid + st]; __syncthreads(); }
  if (tid == 0) sinv = rsqrtf(red[0] / (float)CD + 1e-5f);
  __syncthreads();
  float inv = sinv;
  float* yr = y + row * (long)CD;
  yr[tid]       = a0 * inv * g[tid]       + be[tid];
  yr[tid + 256] = a1 * inv * g[tid + 256] + be[tid + 256];
}

__global__ __launch_bounds__(256)
void amp_k(const float* __restrict__ Re, const float* __restrict__ Im, float* __restrict__ amp)
{
  int bf = blockIdx.x;
  const float* re = Re + (long)bf * CD;
  const float* im = Im + (long)bf * CD;
  float s = 0.f;
  for (int d = threadIdx.x; d < CD; d += 256) s += sqrtf(re[d] * re[d] + im[d] * im[d]);
  __shared__ float red[256];
  red[threadIdx.x] = s; __syncthreads();
  for (int st = 128; st > 0; st >>= 1) { if (threadIdx.x < st) red[threadIdx.x] += red[threadIdx.x + st]; __syncthreads(); }
  if (threadIdx.x == 0) amp[bf] = red[0] / (float)CD;
}

__global__ void topk_k(float* __restrict__ amp, int* __restrict__ idx)
{
  if (threadIdx.x != 0) return;
  int b = blockIdx.x;
  float* a = amp + (long)b * CFQ;
  for (int j = 0; j < CTOPK; ++j) {
    float best = -3.4e38f; int bi = 0;
    for (int f = 0; f < CFQ; ++f) if (a[f] > best) { best = a[f]; bi = f; }
    idx[b * CTOPK + j] = bi;
    a[bi] = -3.4e38f;
  }
}

__global__ void gather_k(const float* __restrict__ Re, const int* __restrict__ idx,
                         float* __restrict__ top)
{
  long i = (long)blockIdx.x * 256 + threadIdx.x;
  const long total = (long)CB * CTOPK * CD;
  if (i >= total) return;
  int d = (int)(i % CD); long r = i / CD;
  int j = (int)(r % CTOPK), b = (int)(r / CTOPK);
  top[i] = Re[((long)b * CFQ + idx[b * CTOPK + j]) * CD + d];
}

__global__ __launch_bounds__(256)
void freq_attn_k(const float* __restrict__ Q, const float* __restrict__ K,
                 const float* __restrict__ V, float* __restrict__ O)
{
  int b = blockIdx.x, tid = threadIdx.x;
  __shared__ float sc[16][17];
  int i = tid >> 4, j = tid & 15;
  const float* q = Q + ((long)b * CTOPK + i) * CD;
  const float* k = K + ((long)b * CTOPK + j) * CD;
  float dot = 0.f;
  for (int d = 0; d < CD; ++d) dot += q[d] * k[d];
  sc[i][j] = dot * 0.04419417382415922f;  // 1/sqrt(512)
  __syncthreads();
  if (tid < 16) {
    float mx = -3.4e38f;
    for (int c = 0; c < 16; ++c) mx = fmaxf(mx, sc[tid][c]);
    float s = 0.f;
    for (int c = 0; c < 16; ++c) { float e = __expf(sc[tid][c] - mx); sc[tid][c] = e; s += e; }
    float inv = 1.f / s;
    for (int c = 0; c < 16; ++c) sc[tid][c] *= inv;
  }
  __syncthreads();
  for (int w = tid; w < CTOPK * CD; w += 256) {
    int ii = w / CD, d = w % CD;
    float s = 0.f;
    for (int c = 0; c < 16; ++c) s += sc[ii][c] * V[((long)b * CTOPK + c) * CD + d];
    O[((long)b * CTOPK + ii) * CD + d] = s;
  }
}

__global__ void irfft_k(const float* __restrict__ O, const int* __restrict__ idx,
                        float* __restrict__ h)
{
  long i = (long)blockIdx.x * 256 + threadIdx.x;
  const long total = (long)CB * CTP * CD;
  if (i >= total) return;
  int d = (int)(i % CD); long r = i / CD;
  int t = (int)(r % CTP), b = (int)(r / CTP);
  float acc = 0.f;
  for (int j = 0; j < CTOPK; ++j) {
    int f = idx[b * CTOPK + j];
    float o = O[((long)b * CTOPK + j) * CD + d];
    int ph = (int)(((long)f * t) % CTP);
    float ang = 6.283185307179586f * (float)ph / (float)CTP;
    float c = cosf(ang), s = sinf(ang);
    if (f == 0 || f == CTP / 2) acc += o * c;          // Re-only bins
    else acc += 2.f * (o * c - o * s);                 // Re=Im=o
  }
  h[i] = acc / (float)CTP;
}

__global__ void pool_k(const float* __restrict__ x, float* __restrict__ pooled)
{
  int i = blockIdx.x * 256 + threadIdx.x;
  if (i >= CB * CD) return;
  int d = i % CD, b = i / CD;
  const float* p = x + (long)b * CTP * CD + d;
  float s = 0.f;
  for (int t = 0; t < CTP; ++t) s += p[(long)t * CD];
  pooled[i] = s / (float)CTP;
}

// ------------------------------- host side ---------------------------------
static void run_gemm(int flags, const float* A, const float* Bm, float* C,
                     const float* bias, const float* resid,
                     int M, int N, int K, long sA, long sB, long sC, long sR,
                     int batch, hipStream_t st)
{
  dim3 g((N + TBN - 1) / TBN, (M + TBM - 1) / TBM, batch), blk(256);
  if (flags == 0)
    gemm_wmma_k<0><<<g, blk, 0, st>>>(A, Bm, C, bias, resid, M, N, K, sA, sB, sC, sR);
  else if (flags == GF_BIAS)
    gemm_wmma_k<GF_BIAS><<<g, blk, 0, st>>>(A, Bm, C, bias, resid, M, N, K, sA, sB, sC, sR);
  else if (flags == (GF_BIAS | GF_RELU))
    gemm_wmma_k<GF_BIAS | GF_RELU><<<g, blk, 0, st>>>(A, Bm, C, bias, resid, M, N, K, sA, sB, sC, sR);
  else
    gemm_wmma_k<GF_BIAS | GF_RESID><<<g, blk, 0, st>>>(A, Bm, C, bias, resid, M, N, K, sA, sB, sC, sR);
}

extern "C" void kernel_launch(void* const* d_in, const int* in_sizes, int n_in,
                              void* d_out, int out_size, void* d_ws, size_t ws_size,
                              hipStream_t stream)
{
  (void)in_sizes; (void)n_in; (void)out_size; (void)ws_size;
  const float* x      = (const float*)d_in[0];
  const float* Win    = (const float*)d_in[1];
  const float* b_in   = (const float*)d_in[2];
  const float* qkv_w  = (const float*)d_in[3];
  const float* qkv_b  = (const float*)d_in[4];
  const float* out_w  = (const float*)d_in[5];
  const float* out_b  = (const float*)d_in[6];
  const float* ff_w1  = (const float*)d_in[7];
  const float* ff_b1  = (const float*)d_in[8];
  const float* ff_w2  = (const float*)d_in[9];
  const float* ff_b2  = (const float*)d_in[10];
  const float* ln1_g  = (const float*)d_in[11];
  const float* ln1_b  = (const float*)d_in[12];
  const float* ln2_g  = (const float*)d_in[13];
  const float* ln2_b  = (const float*)d_in[14];
  const float* fq_w   = (const float*)d_in[15];
  const float* fq_b   = (const float*)d_in[16];
  const float* fk_w   = (const float*)d_in[17];
  const float* fk_b   = (const float*)d_in[18];
  const float* fv_w   = (const float*)d_in[19];
  const float* fv_b   = (const float*)d_in[20];
  const float* head_w = (const float*)d_in[21];
  const float* head_b = (const float*)d_in[22];
  float* outp = (float*)d_out;

  const long NT = (long)CB * CTP;   // 16768
  char* wp = (char*)d_ws;
  auto alloc = [&](long nflt) -> float* {
    float* p = (float*)wp; wp += nflt * sizeof(float); return p;
  };
  float* trend = alloc((long)CB * CTP * CFIN);
  float* seas  = alloc((long)CB * CTP * CFIN);
  float* h     = alloc(NT * CD);
  float* tmp   = alloc(NT * CD);
  float* qkvb  = alloc(NT * 3 * CD);
  float* attnb = alloc(NT * CD);
  float* ffm   = alloc(NT * CDFF);
  float* cosw  = alloc((long)CFQ * CTP);
  float* sinw  = alloc((long)CFQ * CTP);
  float* Re    = alloc((long)CB * CFQ * CD);
  float* Im    = alloc((long)CB * CFQ * CD);
  float* amp   = alloc(CB * CFQ);
  int*   idxb  = (int*)alloc(CB * CTOPK);
  float* topb  = alloc((long)CB * CTOPK * CD);
  float* fqb   = alloc((long)CB * CTOPK * CD);
  float* fkb   = alloc((long)CB * CTOPK * CD);
  float* fvb   = alloc((long)CB * CTOPK * CD);
  float* fob   = alloc((long)CB * CTOPK * CD);
  float* pooled= alloc(CB * CD);

  { long n = (long)CFQ * CTP;
    twiddle_k<<<dim3((unsigned)((n + 255) / 256)), 256, 0, stream>>>(cosw, sinw); }
  { long n = (long)CB * CTP * CFIN;
    decompose_k<<<dim3((unsigned)((n + 255) / 256)), 256, 0, stream>>>(x, trend, seas); }

  // h = seasonal @ Win + b_in + posenc
  run_gemm(GF_BIAS, seas, Win, h, b_in, nullptr, (int)NT, CD, CFIN, 0, 0, 0, 0, 1, stream);
  { long n = NT * (long)CD;
    posenc_add_k<<<dim3((unsigned)((n + 255) / 256)), 256, 0, stream>>>(h); }

  for (int l = 0; l < 2; ++l) {
    run_gemm(GF_BIAS, h, qkv_w + (long)l * CD * 3 * CD, qkvb, qkv_b + l * 3 * CD,
             nullptr, (int)NT, 3 * CD, CD, 0, 0, 0, 0, 1, stream);
    flash_attn_k<<<dim3(CB * CNH, 9), 256, 0, stream>>>(qkvb, attnb);
    run_gemm(GF_BIAS | GF_RESID, attnb, out_w + (long)l * CD * CD, tmp,
             out_b + l * CD, h, (int)NT, CD, CD, 0, 0, 0, 0, 1, stream);
    ln_k<<<dim3((unsigned)NT), 256, 0, stream>>>(tmp, ln1_g + l * CD, ln1_b + l * CD, h);
    run_gemm(GF_BIAS | GF_RELU, h, ff_w1 + (long)l * CD * CDFF, ffm,
             ff_b1 + l * CDFF, nullptr, (int)NT, CDFF, CD, 0, 0, 0, 0, 1, stream);
    run_gemm(GF_BIAS | GF_RESID, ffm, ff_w2 + (long)l * CDFF * CD, tmp,
             ff_b2 + l * CD, h, (int)NT, CD, CDFF, 0, 0, 0, 0, 1, stream);
    ln_k<<<dim3((unsigned)NT), 256, 0, stream>>>(tmp, ln2_g + l * CD, ln2_b + l * CD, h);

    // frequency block: DFT as batched WMMA GEMM
    run_gemm(0, cosw, h, Re, nullptr, nullptr, CFQ, CD, CTP,
             0, (long)CTP * CD, (long)CFQ * CD, 0, CB, stream);
    run_gemm(0, sinw, h, Im, nullptr, nullptr, CFQ, CD, CTP,
             0, (long)CTP * CD, (long)CFQ * CD, 0, CB, stream);
    amp_k<<<dim3(CB * CFQ), 256, 0, stream>>>(Re, Im, amp);
    topk_k<<<dim3(CB), 32, 0, stream>>>(amp, idxb);
    { long n = (long)CB * CTOPK * CD;
      gather_k<<<dim3((unsigned)((n + 255) / 256)), 256, 0, stream>>>(Re, idxb, topb); }
    run_gemm(GF_BIAS, topb, fq_w + (long)l * CD * CD, fqb, fq_b + l * CD, nullptr,
             CB * CTOPK, CD, CD, 0, 0, 0, 0, 1, stream);
    run_gemm(GF_BIAS, topb, fk_w + (long)l * CD * CD, fkb, fk_b + l * CD, nullptr,
             CB * CTOPK, CD, CD, 0, 0, 0, 0, 1, stream);
    run_gemm(GF_BIAS, topb, fv_w + (long)l * CD * CD, fvb, fv_b + l * CD, nullptr,
             CB * CTOPK, CD, CD, 0, 0, 0, 0, 1, stream);
    freq_attn_k<<<dim3(CB), 256, 0, stream>>>(fqb, fkb, fvb, fob);
    { long n = NT * (long)CD;
      irfft_k<<<dim3((unsigned)((n + 255) / 256)), 256, 0, stream>>>(fob, idxb, h); }
  }

  // tmp = h + trend @ Win + b_in ; pool ; head
  run_gemm(GF_BIAS | GF_RESID, trend, Win, tmp, b_in, h, (int)NT, CD, CFIN,
           0, 0, 0, 0, 1, stream);
  pool_k<<<dim3((CB * CD + 255) / 256), 256, 0, stream>>>(tmp, pooled);
  run_gemm(GF_BIAS, pooled, head_w, outp, head_b, nullptr, CB, COUTN, CD,
           0, 0, 0, 0, 1, stream);
}
